// TEN_37177236914700
// MI455X (gfx1250) — matrix operand order repaired and checked by reading, not compiled, not run on hardware
//
#include <hip/hip_runtime.h>

// ---------------------------------------------------------------------------
// Problem constants (from the reference)
// ---------------------------------------------------------------------------
static const int kN0 = 20000, kN1 = 40000, kN2 = 24000;
static const int kNC[3]  = {kN0, kN1, kN2};
static const int kSA[4]  = {0, 0, 1, 1};            // sender rank per adjacency
static const int kRA[4]  = {0, 1, 1, 2};            // receiver rank per adjacency
static const int kNE[4]  = {160000, 80000, 120000, 72000};
static const int kFACTOR[3] = {2, 3, 2};
#define H      128
#define FIN    15
#define LAYERS 2
#define BATCH  64
#define K1EDGE 288   // 2H+3 = 259 padded to multiple of 32

// ---------------------------------------------------------------------------
// WMMA types (CDNA5 / gfx1250, wave32)
// ---------------------------------------------------------------------------
typedef __attribute__((ext_vector_type(16))) __bf16 v16bf;
typedef __attribute__((ext_vector_type(8)))  float  v8f;

union Frag32B { v16bf v; uint4 q[2]; };

static const int LDSTR = 392;   // LDS row stride in halfs: 384 max K + 8 pad (kills bank conflicts)

__device__ inline float siluf(float x) { return x / (1.f + __expf(-x)); }

// A-fragment (16x32 bf16): lane L holds row m=L%16; group g=L/16 selects K halves.
// halfs 0..7  -> K = 32*kt + 8g + (0..7)   (contiguous 16B)
// halfs 8..15 -> K = 32*kt + 16 + 8g + (0..7)
__device__ inline v16bf load_a_frag(const __bf16* row, int ktBase, int g) {
    Frag32B f;
    f.q[0] = *(const uint4*)(row + ktBase + 8 * g);
    f.q[1] = *(const uint4*)(row + ktBase + 16 + 8 * g);
    return f.v;
}

// B-fragment: pre-packed contiguously per (ktile, ntile, lane): 16 halfs = 32B.
__device__ inline v16bf load_b_frag(const __bf16* p) {
    Frag32B f;
    f.q[0] = *(const uint4*)(p);
    f.q[1] = *(const uint4*)(p + 8);
    return f.v;
}

// ---------------------------------------------------------------------------
// Weight pre-pack: f32 [K,128] row-major -> bf16 WMMA B-fragment layout,
// K zero-padded to Kp (multiple of 32).
// Fragment (kt,nt): half i holds K = 32*kt + 8*(lane/16) + (i%8) + 16*(i/8), N = 16*nt + lane%16.
// ---------------------------------------------------------------------------
__global__ void pack_b_kernel(const float* __restrict__ W, __bf16* __restrict__ P,
                              int K, int Kp) {
    int tid = blockIdx.x * 256 + threadIdx.x;
    int total = Kp * 128;                 // (Kp/32)*8 ntiles*32 lanes*16 halfs
    if (tid >= total) return;
    int i    = tid & 15;
    int lane = (tid >> 4) & 31;
    int rest = tid >> 9;
    int nt   = rest & 7;
    int kt   = rest >> 3;
    int g = lane >> 4;
    int n = nt * 16 + (lane & 15);
    int k = kt * 32 + 8 * g + (i & 7) + 16 * (i >> 3);
    float v = (k < K) ? W[(size_t)k * 128 + n] : 0.f;
    P[tid] = (__bf16)v;
}

// ---------------------------------------------------------------------------
// Initial features: x[d] = emb(mean over cell vertices of x_node)
// ---------------------------------------------------------------------------
__global__ void init_feats_kernel(const float* __restrict__ xnode,
                                  const int* __restrict__ xind, int nv, int nc,
                                  const float* __restrict__ W,
                                  const float* __restrict__ bias,
                                  float* __restrict__ xout) {
    int gid = blockIdx.x * 256 + threadIdx.x;
    if (gid >= nc * H) return;
    int cell = gid >> 7, h = gid & 127;
    float xm[FIN];
#pragma unroll
    for (int k = 0; k < FIN; ++k) xm[k] = 0.f;
    for (int v = 0; v < nv; ++v) {
        int vi = xind[cell * nv + v];
#pragma unroll
        for (int k = 0; k < FIN; ++k) xm[k] += xnode[(size_t)vi * FIN + k];
    }
    float s = 1.f / (float)nv;
    float acc = bias[h];
#pragma unroll
    for (int k = 0; k < FIN; ++k) acc += xm[k] * s * W[k * H + h];
    xout[(size_t)cell * H + h] = acc;
}

// ---------------------------------------------------------------------------
// Geometric invariants per edge: [centroid dist, directed Hausdorff s->r, r->s]
// ---------------------------------------------------------------------------
__global__ void invariants_kernel(const int* __restrict__ adj, int E,
                                  const int* __restrict__ xind_s, int ns,
                                  const int* __restrict__ xind_r, int nr,
                                  const float* __restrict__ pos,
                                  float* __restrict__ invout) {
    int e = blockIdx.x * 256 + threadIdx.x;
    if (e >= E) return;
    int s = adj[e], r = adj[E + e];
    float ps[3][3], pr[3][3], cs[3] = {0, 0, 0}, cr[3] = {0, 0, 0};
    for (int v = 0; v < ns; ++v) {
        int vi = xind_s[s * ns + v];
        for (int c = 0; c < 3; ++c) { ps[v][c] = pos[(size_t)vi * 3 + c]; cs[c] += ps[v][c]; }
    }
    for (int v = 0; v < nr; ++v) {
        int vi = xind_r[r * nr + v];
        for (int c = 0; c < 3; ++c) { pr[v][c] = pos[(size_t)vi * 3 + c]; cr[c] += pr[v][c]; }
    }
    float d2 = 0.f;
    for (int c = 0; c < 3; ++c) {
        float d = cs[c] / (float)ns - cr[c] / (float)nr;
        d2 += d * d;
    }
    float cd = sqrtf(d2 + 1e-12f);
    float dm[3][3];
    for (int i = 0; i < ns; ++i)
        for (int j = 0; j < nr; ++j) {
            float a = 0.f;
            for (int c = 0; c < 3; ++c) { float d = ps[i][c] - pr[j][c]; a += d * d; }
            dm[i][j] = sqrtf(a + 1e-12f);
        }
    float h1 = -1e30f;
    for (int i = 0; i < ns; ++i) {
        float m = 1e30f;
        for (int j = 0; j < nr; ++j) m = fminf(m, dm[i][j]);
        h1 = fmaxf(h1, m);
    }
    float h2 = -1e30f;
    for (int j = 0; j < nr; ++j) {
        float m = 1e30f;
        for (int i = 0; i < ns; ++i) m = fminf(m, dm[i][j]);
        h2 = fmaxf(h2, m);
    }
    invout[(size_t)e * 3 + 0] = cd;
    invout[(size_t)e * 3 + 1] = h1;
    invout[(size_t)e * 3 + 2] = h2;
}

// ---------------------------------------------------------------------------
// Stage a (possibly gathered) f32 source into the per-wave LDS tile as bf16.
// nRows = 32 rows per wave tile.
// ---------------------------------------------------------------------------
__device__ inline void stage_src(__bf16* As, const float* __restrict__ src,
                                 const int* __restrict__ idx, int w, int colOff,
                                 int rowBase, int M, int lane) {
    if ((w & 3) == 0) {
        int nv = w >> 2;
        for (int t = lane; t < 32 * nv; t += 32) {
            int r = t / nv, c = (t - r * nv) << 2;
            __bf16* dst = As + r * LDSTR + colOff + c;
            union { __bf16 h[4]; uint2 u; } pk;
            int row = rowBase + r;
            if (row < M) {
                int ri = idx ? idx[row] : row;
                float4 f = *(const float4*)(src + (size_t)ri * w + c);
                pk.h[0] = (__bf16)f.x; pk.h[1] = (__bf16)f.y;
                pk.h[2] = (__bf16)f.z; pk.h[3] = (__bf16)f.w;
            } else {
                pk.u = make_uint2(0u, 0u);
            }
            *(uint2*)dst = pk.u;
        }
    } else {
        for (int t = lane; t < 32 * w; t += 32) {
            int r = t / w, c = t - r * w;
            int row = rowBase + r;
            float v = 0.f;
            if (row < M) { int ri = idx ? idx[row] : row; v = src[(size_t)ri * w + c]; }
            As[r * LDSTR + colOff + c] = (__bf16)v;
        }
    }
}

// ---------------------------------------------------------------------------
// Fused 2-layer MLP over 32-row tiles via v_wmma_f32_16x16x32_bf16.
// Each wave owns two 16-row sub-tiles that SHARE every B-fragment load
// (halves weight traffic per MAC -> keeps the matrix pipe fed from WGP$).
//   A = concat(gather(src0), gather(src1), gather(src2)) zero-padded to K1p
//   hM = silu(A @ W1 + b1)         (K1p/32 x 8 x 2 WMMAs)
//   y  = h @ W2 + b2               (4 x 8 x 2 WMMAs)
//   mode 0: y=silu(y); gate=sigmoid(y.We+be); atomicAdd y*gate into out[idx_r]
//   mode 1: out[row] = resid[row] + y          (residual cell update, in place)
//   mode 2: atomicAdd y into out[batch[row]*outStride + outColOff]  (global pool)
// 2 waves/block, per-wave LDS slice, no block-wide barriers.
// ---------------------------------------------------------------------------
__global__ __launch_bounds__(64)
void fused_mlp_kernel(const float* __restrict__ src0, const int* __restrict__ idx0, int w0,
                      const float* __restrict__ src1, const int* __restrict__ idx1, int w1,
                      const float* __restrict__ src2, const int* __restrict__ idx2, int w2,
                      int M, int K1p,
                      const __bf16* __restrict__ W1p, const float* __restrict__ b1,
                      const __bf16* __restrict__ W2p, const float* __restrict__ b2,
                      int mode,
                      const float* __restrict__ gateW, const float* __restrict__ gateB,
                      const int* __restrict__ scatterIdx,
                      const float* __restrict__ resid,
                      float* __restrict__ out, int outStride, int outColOff) {
    __shared__ __bf16 smem[2][32 * LDSTR];
    int wave = threadIdx.x >> 5, lane = threadIdx.x & 31;
    int rowBase = (blockIdx.x * 2 + wave) * 32;
    if (rowBase >= M) return;                 // safe: no block-wide barriers used
    __bf16* As = smem[wave];

    // ---- stage A tile (32 rows, bf16) ----
    stage_src(As, src0, idx0, w0, 0, rowBase, M, lane);
    int colOff = w0;
    if (src1) { stage_src(As, src1, idx1, w1, colOff, rowBase, M, lane); colOff += w1; }
    if (src2) { stage_src(As, src2, idx2, w2, colOff, rowBase, M, lane); colOff += w2; }
    int pad = K1p - colOff;
    if (pad > 0) {
        for (int t = lane; t < 32 * pad; t += 32) {
            int r = t / pad, c = colOff + (t - r * pad);
            As[r * LDSTR + c] = (__bf16)0.f;
        }
    }
    asm volatile("s_wait_dscnt 0" ::: "memory");

    int g = lane >> 4, ln = lane & 15;
    const __bf16* arow0 = As + ln * LDSTR;            // sub-tile 0 (rows 0..15)
    const __bf16* arow1 = As + (16 + ln) * LDSTR;     // sub-tile 1 (rows 16..31)

    // ---- GEMM1: [32,K1p] x [K1p,128], B fragments shared by both sub-tiles ----
    v8f c1[2][8] = {};
    int nkt = K1p >> 5;
    for (int kt = 0; kt < nkt; ++kt) {
        v16bf a0 = load_a_frag(arow0, kt * 32, g);
        v16bf a1 = load_a_frag(arow1, kt * 32, g);
        const __bf16* bp = W1p + (size_t)(kt * 8) * 512 + lane * 16;
#pragma unroll
        for (int nt = 0; nt < 8; ++nt) {
            v16bf b = load_b_frag(bp + (size_t)nt * 512);
            c1[0][nt] = __builtin_amdgcn_wmma_f32_16x16x32_bf16(
                false, a0, false, b, (short)0, c1[0][nt], false, false);
            c1[1][nt] = __builtin_amdgcn_wmma_f32_16x16x32_bf16(
                false, a1, false, b, (short)0, c1[1][nt], false, false);
        }
    }

    // ---- bias + silu, round-trip h through per-wave LDS as bf16 ----
#pragma unroll
    for (int nt = 0; nt < 8; ++nt) {
        float bb = b1[nt * 16 + ln];
#pragma unroll
        for (int s = 0; s < 2; ++s)
#pragma unroll
            for (int r = 0; r < 8; ++r) {
                float v = siluf(c1[s][nt][r] + bb);
                As[(s * 16 + r + 8 * g) * LDSTR + nt * 16 + ln] = (__bf16)v;
            }
    }
    asm volatile("s_wait_dscnt 0" ::: "memory");

    // ---- GEMM2: [32,128] x [128,128] ----
    v8f c2[2][8] = {};
#pragma unroll
    for (int kt = 0; kt < 4; ++kt) {
        v16bf a0 = load_a_frag(arow0, kt * 32, g);
        v16bf a1 = load_a_frag(arow1, kt * 32, g);
        const __bf16* bp = W2p + (size_t)(kt * 8) * 512 + lane * 16;
#pragma unroll
        for (int nt = 0; nt < 8; ++nt) {
            v16bf b = load_b_frag(bp + (size_t)nt * 512);
            c2[0][nt] = __builtin_amdgcn_wmma_f32_16x16x32_bf16(
                false, a0, false, b, (short)0, c2[0][nt], false, false);
            c2[1][nt] = __builtin_amdgcn_wmma_f32_16x16x32_bf16(
                false, a1, false, b, (short)0, c2[1][nt], false, false);
        }
    }

    if (mode == 0) {
        // edge path: y = silu(y+b2); w = sigmoid(y.We + be); scatter-add y*w
#pragma unroll
        for (int nt = 0; nt < 8; ++nt) {
            float bb = b2[nt * 16 + ln];
#pragma unroll
            for (int s = 0; s < 2; ++s)
#pragma unroll
                for (int r = 0; r < 8; ++r) c2[s][nt][r] = siluf(c2[s][nt][r] + bb);
        }
        float dot[2][8];
#pragma unroll
        for (int s = 0; s < 2; ++s)
#pragma unroll
            for (int r = 0; r < 8; ++r) dot[s][r] = 0.f;
#pragma unroll
        for (int nt = 0; nt < 8; ++nt) {
            float wv = gateW[nt * 16 + ln];
#pragma unroll
            for (int s = 0; s < 2; ++s)
#pragma unroll
                for (int r = 0; r < 8; ++r) dot[s][r] += c2[s][nt][r] * wv;
        }
#pragma unroll
        for (int s = 0; s < 2; ++s)
#pragma unroll
            for (int r = 0; r < 8; ++r)
                for (int o = 1; o < 16; o <<= 1) dot[s][r] += __shfl_xor(dot[s][r], o, 16);
        float gb = gateB[0];
#pragma unroll
        for (int s = 0; s < 2; ++s)
#pragma unroll
            for (int r = 0; r < 8; ++r) {
                int row = rowBase + s * 16 + r + 8 * g;
                if (row < M) {
                    float wgt = 1.f / (1.f + __expf(-(dot[s][r] + gb)));
                    int ridx = scatterIdx[row];
                    float* op = out + (size_t)ridx * outStride;
#pragma unroll
                    for (int nt = 0; nt < 8; ++nt)
                        atomicAdd(op + nt * 16 + ln, c2[s][nt][r] * wgt);
                }
            }
    } else if (mode == 1) {
        // residual cell update (row-exclusive, safe in place)
#pragma unroll
        for (int s = 0; s < 2; ++s)
#pragma unroll
            for (int r = 0; r < 8; ++r) {
                int row = rowBase + s * 16 + r + 8 * g;
                if (row < M) {
#pragma unroll
                    for (int nt = 0; nt < 8; ++nt) {
                        int col = nt * 16 + ln;
                        out[(size_t)row * H + col] =
                            resid[(size_t)row * H + col] + c2[s][nt][r] + b2[col];
                    }
                }
            }
    } else {
        // pre-pool: segment-sum by batch index into pooled[B, 384]
#pragma unroll
        for (int s = 0; s < 2; ++s)
#pragma unroll
            for (int r = 0; r < 8; ++r) {
                int row = rowBase + s * 16 + r + 8 * g;
                if (row < M) {
                    int bidx = scatterIdx[row];
                    float* op = out + (size_t)bidx * outStride + outColOff;
#pragma unroll
                    for (int nt = 0; nt < 8; ++nt) {
                        int col = nt * 16 + ln;
                        atomicAdd(op + col, c2[s][nt][r] + b2[col]);
                    }
                }
            }
    }
}

// ---------------------------------------------------------------------------
// Post MLP: out[b] = (silu(pooled[b]@W1+b1)) @ W2 + b2       (B=64 tiny rows)
// ---------------------------------------------------------------------------
__global__ void post_kernel(const float* __restrict__ pooled,
                            const float* __restrict__ W1, const float* __restrict__ b1,
                            const float* __restrict__ W2, const float* __restrict__ b2,
                            float* __restrict__ out) {
    __shared__ float sh[H];
    int b = blockIdx.x, t = threadIdx.x;
    float acc = b1[t];
    for (int k = 0; k < 3 * H; ++k) acc += pooled[(size_t)b * 3 * H + k] * W1[k * H + t];
    sh[t] = siluf(acc) * W2[t];
    __syncthreads();
    for (int s = 64; s > 0; s >>= 1) {
        if (t < s) sh[t] += sh[t + s];
        __syncthreads();
    }
    if (t == 0) out[b] = sh[0] + b2[0];
}

// ---------------------------------------------------------------------------
// Host orchestration
// ---------------------------------------------------------------------------
// Input flattening assumption: setup_inputs() insertion order, with the nested
// `params` dict flattened in make_params() insertion order as [W, b] leaves:
//   0: x_node   1: pos   2..91: params (90 leaves)   92..94: x_ind_{0,1,2}
//   95..98: adj_{0_0,0_1,1_1,1_2}   99..101: batch_{0,1,2}   102: batch_size
#define P0 2
#define MI(l, a, j)  (P0 + 2 + (l) * 36 + (a) * 6 + (j))   // m1W,m1b,m2W,m2b,eW,eb
#define UI(l, d, j)  (P0 + 2 + (l) * 36 + 24 + (d) * 4 + (j)) // u1W,u1b,u2W,u2b
#define PREI(d, j)   (P0 + 2 + 72 + (d) * 4 + (j))
#define POSTI(j)     (P0 + 2 + 72 + 12 + (j))
#define IN_XIND(d)   (92 + (d))
#define IN_ADJ(a)    (95 + (a))
#define IN_BATCH(d)  (99 + (d))

extern "C" void kernel_launch(void* const* d_in, const int* in_sizes, int n_in,
                              void* d_out, int out_size, void* d_ws, size_t ws_size,
                              hipStream_t stream) {
    (void)in_sizes; (void)n_in; (void)out_size; (void)ws_size;
    auto F = [&](int i) { return (const float*)d_in[i]; };
    auto I = [&](int i) { return (const int*)d_in[i]; };

    // ---- workspace bump allocator ----
    char* wsb = (char*)d_ws;
    size_t off = 0;
    auto alloc = [&](size_t bytes) -> void* {
        void* p = wsb + off;
        off = (off + bytes + 255) & ~(size_t)255;
        return p;
    };
    float* xf[3];
    for (int d = 0; d < 3; ++d) xf[d] = (float*)alloc((size_t)kNC[d] * H * 4);
    float* invb[4];
    for (int a = 0; a < 4; ++a) invb[a] = (float*)alloc((size_t)kNE[a] * 3 * 4);
    float* mes[4];
    size_t mesB[4];
    for (int a = 0; a < 4; ++a) {
        mesB[a] = (size_t)kNC[kRA[a]] * H * 4;
        mes[a] = (float*)alloc(mesB[a]);
    }
    __bf16 *m1p[LAYERS][4], *m2p[LAYERS][4], *u1p[LAYERS][3], *u2p[LAYERS][3], *p1p[3], *p2p[3];
    for (int l = 0; l < LAYERS; ++l)
        for (int a = 0; a < 4; ++a) {
            m1p[l][a] = (__bf16*)alloc((size_t)K1EDGE * H * 2);
            m2p[l][a] = (__bf16*)alloc((size_t)H * H * 2);
        }
    for (int l = 0; l < LAYERS; ++l)
        for (int d = 0; d < 3; ++d) {
            u1p[l][d] = (__bf16*)alloc((size_t)kFACTOR[d] * H * H * 2);
            u2p[l][d] = (__bf16*)alloc((size_t)H * H * 2);
        }
    for (int d = 0; d < 3; ++d) {
        p1p[d] = (__bf16*)alloc((size_t)H * H * 2);
        p2p[d] = (__bf16*)alloc((size_t)H * H * 2);
    }
    float* pooled = (float*)alloc((size_t)BATCH * 3 * H * 4);

    auto pack = [&](const float* W, __bf16* P, int K, int Kp) {
        int total = Kp * H;
        pack_b_kernel<<<(total + 255) / 256, 256, 0, stream>>>(W, P, K, Kp);
    };

    // ---- pre-swizzle all GEMM weights to bf16 WMMA-B layout ----
    for (int l = 0; l < LAYERS; ++l)
        for (int a = 0; a < 4; ++a) {
            pack(F(MI(l, a, 0)), m1p[l][a], 2 * H + 3, K1EDGE);
            pack(F(MI(l, a, 2)), m2p[l][a], H, H);
        }
    for (int l = 0; l < LAYERS; ++l)
        for (int d = 0; d < 3; ++d) {
            pack(F(UI(l, d, 0)), u1p[l][d], kFACTOR[d] * H, kFACTOR[d] * H);
            pack(F(UI(l, d, 2)), u2p[l][d], H, H);
        }
    for (int d = 0; d < 3; ++d) {
        pack(F(PREI(d, 0)), p1p[d], H, H);
        pack(F(PREI(d, 2)), p2p[d], H, H);
    }

    // ---- initial embedded features ----
    for (int d = 0; d < 3; ++d) {
        int n = kNC[d] * H;
        init_feats_kernel<<<(n + 255) / 256, 256, 0, stream>>>(
            F(0), I(IN_XIND(d)), d + 1, kNC[d], F(P0), F(P0 + 1), xf[d]);
    }

    // ---- geometric invariants ----
    for (int a = 0; a < 4; ++a) {
        invariants_kernel<<<(kNE[a] + 255) / 256, 256, 0, stream>>>(
            I(IN_ADJ(a)), kNE[a],
            I(IN_XIND(kSA[a])), kSA[a] + 1,
            I(IN_XIND(kRA[a])), kRA[a] + 1,
            F(1), invb[a]);
    }

    auto fused = [&](const float* s0, const int* i0, int w0,
                     const float* s1, const int* i1, int w1,
                     const float* s2, const int* i2, int w2,
                     int M, int K1p, const __bf16* W1, const float* B1,
                     const __bf16* W2, const float* B2, int mode,
                     const float* gw, const float* gb, const int* sc,
                     const float* rs, float* o, int os, int oc) {
        int grid = (M + 63) / 64;   // 2 waves/block * 32 rows/wave
        fused_mlp_kernel<<<grid, 64, 0, stream>>>(
            s0, i0, w0, s1, i1, w1, s2, i2, w2, M, K1p,
            W1, B1, W2, B2, mode, gw, gb, sc, rs, o, os, oc);
    };

    // ---- message-passing layers ----
    for (int l = 0; l < LAYERS; ++l) {
        for (int a = 0; a < 4; ++a) hipMemsetAsync(mes[a], 0, mesB[a], stream);
        for (int a = 0; a < 4; ++a) {
            const int* adj = I(IN_ADJ(a));
            fused(xf[kSA[a]], adj, H,                 // sender features (gathered)
                  xf[kRA[a]], adj + kNE[a], H,        // receiver features (gathered)
                  invb[a], nullptr, 3,                // geometric invariants
                  kNE[a], K1EDGE,
                  m1p[l][a], F(MI(l, a, 1)),
                  m2p[l][a], F(MI(l, a, 3)),
                  /*mode=*/0, F(MI(l, a, 4)), F(MI(l, a, 5)),
                  adj + kNE[a], nullptr, mes[a], H, 0);
        }
        for (int d = 0; d < 3; ++d) {
            const float* s1 = (d == 0) ? mes[0] : (d == 1) ? mes[1] : mes[3];
            const float* s2 = (d == 1) ? mes[2] : nullptr;
            fused(xf[d], nullptr, H,
                  s1, nullptr, H,
                  s2, nullptr, s2 ? H : 0,
                  kNC[d], kFACTOR[d] * H,
                  u1p[l][d], F(UI(l, d, 1)),
                  u2p[l][d], F(UI(l, d, 3)),
                  /*mode=*/1, nullptr, nullptr, nullptr,
                  xf[d], xf[d], H, 0);   // row-exclusive in-place residual update
        }
    }

    // ---- pre-pool MLP + global add pool ----
    hipMemsetAsync(pooled, 0, (size_t)BATCH * 3 * H * 4, stream);
    for (int d = 0; d < 3; ++d) {
        fused(xf[d], nullptr, H,
              nullptr, nullptr, 0, nullptr, nullptr, 0,
              kNC[d], H,
              p1p[d], F(PREI(d, 1)),
              p2p[d], F(PREI(d, 3)),
              /*mode=*/2, nullptr, nullptr, I(IN_BATCH(d)),
              nullptr, pooled, 3 * H, d * H);
    }

    // ---- post MLP ----
    post_kernel<<<BATCH, H, 0, stream>>>(pooled, F(POSTI(0)), F(POSTI(1)),
                                         F(POSTI(2)), F(POSTI(3)), (float*)d_out);
}
